// GNN_76098230550537
// MI455X (gfx1250) — matrix operand order repaired and checked by reading, not compiled, not run on hardware
//
#include <hip/hip_runtime.h>
#include <cstdint>
#include <cstddef>

// ---------------------------------------------------------------- types ----
typedef __attribute__((ext_vector_type(16))) __bf16  v16bf;
typedef __attribute__((ext_vector_type(8)))  float   v8f;
typedef __attribute__((ext_vector_type(4)))  unsigned int u32x4;

union FragAB { v16bf v; u32x4 q[2]; };

#define HID 64
#define WN  128          // edge-MLP width (2*H)
#define HH  4096         // H*H
#define NTILES 256       // 4096 / 16
#define EPB 32           // edges per block in fused layer kernel (2 M-tiles)

__device__ __forceinline__ unsigned short f2bf(float f) {
    unsigned u = __float_as_uint(f);
    u = (u + 0x7FFFu + ((u >> 16) & 1u)) >> 16;
    return (unsigned short)u;
}
__device__ __forceinline__ float sigmoidf_(float x) { return 1.0f / (1.0f + __expf(-x)); }
__device__ __forceinline__ float siluf_(float x)    { return x * sigmoidf_(x); }

__device__ __forceinline__ void atomicMaxFloat(float* addr, float val) {
    if (val >= 0.0f) atomicMax((int*)addr, __float_as_int(val));
    else             atomicMin((unsigned int*)addr, __float_as_uint(val));
}

// ------------------------------------------------------------- kernels ----
__global__ void zero_f32(float* p, int n) {
    int g = blockIdx.x * blockDim.x + threadIdx.x;
    if (g < n) p[g] = 0.0f;
}

// x (N,11) @ fl_w (11,64) + b, SiLU -> state (N,64)
__global__ void first_layer(const float* __restrict__ x, const float* __restrict__ w,
                            const float* __restrict__ b, float* __restrict__ state, int nN) {
    int g = blockIdx.x * blockDim.x + threadIdx.x;
    if (g >= nN * HID) return;
    int n = g >> 6, o = g & 63;
    float acc = b[o];
    const float* xr = x + n * 11;
#pragma unroll
    for (int k = 0; k < 11; ++k) acc += xr[k] * w[k * HID + o];
    state[g] = siluf_(acc);
}

// hidden = bf16( silu(edge_attr (E,4) @ nn_w1 (4,128) + b1) )
__global__ void edge_mlp1(const float* __restrict__ ea, const float* __restrict__ w1,
                          const float* __restrict__ b1, unsigned short* __restrict__ hidden, int nE) {
    int g = blockIdx.x * blockDim.x + threadIdx.x;
    if (g >= nE * WN) return;
    int e = g >> 7, w = g & 127;
    const float* a = ea + e * 4;
    float acc = b1[w];
#pragma unroll
    for (int k = 0; k < 4; ++k) acc += a[k] * w1[k * WN + w];
    hidden[g] = f2bf(siluf_(acc));
}

// Pack nn_w2 (128,4096) f32 into WMMA-B fragment order (bf16):
// p = ((ntile*4 + kblk)*32 + lane)*16 + i ;  K = kblk*32 + (i>>3)*16 + (lane>>4)*8 + (i&7) ; col = ntile*16 + (lane&15)
__global__ void pack_w2(const float* __restrict__ w2, unsigned short* __restrict__ w2p) {
    int p = blockIdx.x * blockDim.x + threadIdx.x;
    if (p >= WN * HH) return;
    int i    = p & 15;
    int lane = (p >> 4) & 31;
    int kblk = (p >> 9) & 3;
    int nt   = p >> 11;
    int K    = kblk * 32 + (i >> 3) * 16 + (lane >> 4) * 8 + (i & 7);
    int col  = nt * 16 + (lane & 15);
    w2p[p] = f2bf(w2[(size_t)K * HH + col]);
}

// Fused NNConv message pass for one layer.
// For N-tile nt, cols nt*16+j map to a SINGLE h = nt>>2 and o = (nt&3)*16+j, so
// the per-edge matvec  msg[m',o] += x[m',h] * theta[m', h*64+o]  is fused directly
// onto the WMMA accumulator: 1 LDS broadcast + 1 FMA per D element. No theta tile.
__global__ void __launch_bounds__(256)
layer_fused(const unsigned short* __restrict__ hidden, const unsigned short* __restrict__ w2p,
            const float* __restrict__ nn_b2, const float* __restrict__ state,
            const int* __restrict__ src, const int* __restrict__ dst,
            float* __restrict__ agg, int nE) {
    extern __shared__ char smem[];
    float* xs     = (float*)smem;                       // 32 edges x 64 feats
    float* msgbuf = xs + EPB * HID;                     // 32 edges x 64 outs

    const int tid  = threadIdx.x;
    const int lane = tid & 31;
    const int wave = tid >> 5;
    const int e0   = blockIdx.x * EPB;

    // init: gather x[src] rows (clamped) and zero the message buffer
    for (int i = tid; i < EPB * HID; i += 256) {
        int el = i >> 6, f = i & 63;
        int ee = e0 + el; if (ee >= nE) ee = nE - 1;
        xs[i] = state[(size_t)src[ee] * HID + f];
        msgbuf[i] = 0.0f;
    }

    // A fragments for two 16-edge M-tiles, all 4 K-blocks (register resident)
    const int m  = lane & 15;
    const int kg = lane >> 4;
    int r0i = e0 + m;      if (r0i >= nE) r0i = nE - 1;
    int r1i = e0 + 16 + m; if (r1i >= nE) r1i = nE - 1;
    FragAB a0[4], a1[4];
    {
        const unsigned short* ar0 = hidden + (size_t)r0i * WN + kg * 8;
        const unsigned short* ar1 = hidden + (size_t)r1i * WN + kg * 8;
#pragma unroll
        for (int kb = 0; kb < 4; ++kb) {
            a0[kb].q[0] = *(const u32x4*)(ar0 + kb * 32);
            a0[kb].q[1] = *(const u32x4*)(ar0 + kb * 32 + 16);
            a1[kb].q[0] = *(const u32x4*)(ar1 + kb * 32);
            a1[kb].q[1] = *(const u32x4*)(ar1 + kb * 32 + 16);
        }
    }
    __syncthreads();

    const int cit = lane & 15;
    // wave w handles obase = (w&3)*16 (constant: nt stride 8 keeps nt&3 fixed)
    // and h = nt>>2 covering one parity class; waves w and w+4 pair up via msgbuf.
    float macc0[8], macc1[8];
#pragma unroll
    for (int r = 0; r < 8; ++r) { macc0[r] = 0.0f; macc1[r] = 0.0f; }
    const int obase = (wave & 3) * 16;
    const int mb0 = kg * 8;

    for (int nt = wave; nt < NTILES; nt += 8) {
        const int h   = nt >> 2;
        const int col = nt * 16 + cit;
        const float bias = nn_b2[col];
        // preload all 4 B fragments (one clause of 8 b128 loads)
        FragAB b[4];
#pragma unroll
        for (int kb = 0; kb < 4; ++kb) {
            const unsigned short* q = w2p + (((size_t)nt * 4 + kb) * 32 + lane) * 16;
            b[kb].q[0] = ((const u32x4*)q)[0];
            b[kb].q[1] = ((const u32x4*)q)[1];
        }
        v8f c0 = {bias, bias, bias, bias, bias, bias, bias, bias};
        v8f c1 = c0;
#pragma unroll
        for (int kb = 0; kb < 4; ++kb)
            c0 = __builtin_amdgcn_wmma_f32_16x16x32_bf16(
                     false, a0[kb].v, false, b[kb].v, (short)0, c0, false, false);
#pragma unroll
        for (int kb = 0; kb < 4; ++kb)
            c1 = __builtin_amdgcn_wmma_f32_16x16x32_bf16(
                     false, a1[kb].v, false, b[kb].v, (short)0, c1, false, false);
        // fuse per-edge matvec: msg[m',obase+cit] += x[m',h] * theta-tile
#pragma unroll
        for (int r = 0; r < 8; ++r) {
            macc0[r] += xs[(mb0 + r) * HID + h] * c0[r];        // LDS broadcast
            macc1[r] += xs[(16 + mb0 + r) * HID + h] * c1[r];
        }
    }

    // combine the two h-parity waves per (m', o) slot
#pragma unroll
    for (int r = 0; r < 8; ++r) {
        atomicAdd(&msgbuf[(mb0 + r) * HID + obase + cit],      macc0[r]);
        atomicAdd(&msgbuf[(16 + mb0 + r) * HID + obase + cit], macc1[r]);
    }
    __syncthreads();

    // scatter-add to destination nodes (guard the clamped tail edges)
    for (int i = tid; i < EPB * HID; i += 256) {
        int el = i >> 6;
        if (e0 + el < nE)
            atomicAdd(&agg[(size_t)dst[e0 + el] * HID + (i & 63)], msgbuf[i]);
    }
}

// m = silu(agg + x@root_w + conv_b); GRU(h=x): state <- h_new  (in place; x==h invariant)
__global__ void __launch_bounds__(256)
node_update(float* __restrict__ state, const float* __restrict__ agg,
            const float* __restrict__ root_w, const float* __restrict__ conv_b,
            const float* __restrict__ w_ih, const float* __restrict__ w_hh,
            const float* __restrict__ b_ih, const float* __restrict__ b_hh, int nN) {
    extern __shared__ float wsm[];
    float* rw  = wsm;                 // 64x64, [h*64+o]
    float* wih = wsm + 4096;          // transposed: [h*192 + j]
    float* whh = wih + 12288;
    const int tid = threadIdx.x;
    for (int i = tid; i < 4096; i += 256) rw[i] = root_w[i];
    for (int i = tid; i < 12288; i += 256) {
        int j = i >> 6, h2 = i & 63;
        wih[h2 * 192 + j] = w_ih[i];
        whh[h2 * 192 + j] = w_hh[i];
    }
    __syncthreads();

    const int lane = tid & 31, wave = tid >> 5;
    const int n = blockIdx.x * 8 + wave;
    if (n >= nN) return;

    const float* xrow = state + (size_t)n * HID;
    float x0 = xrow[lane], x1 = xrow[lane + 32];

    float acc0 = conv_b[lane]      + agg[(size_t)n * HID + lane];
    float acc1 = conv_b[lane + 32] + agg[(size_t)n * HID + lane + 32];
    for (int h = 0; h < HID; ++h) {
        float xh = (h < 32) ? __shfl(x0, h, 32) : __shfl(x1, h - 32, 32);
        acc0 += xh * rw[h * HID + lane];
        acc1 += xh * rw[h * HID + lane + 32];
    }
    float m0 = siluf_(acc0), m1 = siluf_(acc1);

    float gi[6], gh[6];
#pragma unroll
    for (int k = 0; k < 6; ++k) { int j = lane + 32 * k; gi[k] = b_ih[j]; gh[k] = b_hh[j]; }
    for (int h = 0; h < HID; ++h) {
        float mh = (h < 32) ? __shfl(m0, h, 32) : __shfl(m1, h - 32, 32);
        float hh = (h < 32) ? __shfl(x0, h, 32) : __shfl(x1, h - 32, 32);
        const float* wi = wih + h * 192;
        const float* wh = whh + h * 192;
#pragma unroll
        for (int k = 0; k < 6; ++k) {
            int j = lane + 32 * k;
            gi[k] += mh * wi[j];
            gh[k] += hh * wh[j];
        }
    }
    float r0 = sigmoidf_(gi[0] + gh[0]);
    float z0 = sigmoidf_(gi[2] + gh[2]);
    float n0 = tanhf(gi[4] + r0 * gh[4]);
    float r1 = sigmoidf_(gi[1] + gh[1]);
    float z1 = sigmoidf_(gi[3] + gh[3]);
    float n1 = tanhf(gi[5] + r1 * gh[5]);
    state[(size_t)n * HID + lane]      = (1.0f - z0) * n0 + z0 * x0;
    state[(size_t)n * HID + lane + 32] = (1.0f - z1) * n1 + z1 * x1;
}

// ---- Set2Set ----
__global__ void __launch_bounds__(256)
lstm_step(const float* __restrict__ q_star, float* __restrict__ hl, float* __restrict__ cl,
          const float* __restrict__ w_ih, const float* __restrict__ w_hh,
          const float* __restrict__ b_ih, const float* __restrict__ b_hh,
          float* __restrict__ emax, float* __restrict__ denom, float* __restrict__ racc) {
    __shared__ float g[256];
    const int b = blockIdx.x, j = threadIdx.x;
    float acc = b_ih[j] + b_hh[j];
    const float* q = q_star + b * 128;
    for (int k = 0; k < 128; ++k) acc += q[k] * w_ih[j * 128 + k];
    const float* hp = hl + b * 64;
    for (int k = 0; k < 64; ++k) acc += hp[k] * w_hh[j * 64 + k];
    g[j] = acc;
    __syncthreads();
    if (j < 64) {
        float i_ = sigmoidf_(g[j]);
        float f_ = sigmoidf_(g[64 + j]);
        float g_ = tanhf(g[128 + j]);
        float o_ = sigmoidf_(g[192 + j]);
        float c = f_ * cl[b * 64 + j] + i_ * g_;
        cl[b * 64 + j] = c;
        hl[b * 64 + j] = o_ * tanhf(c);
        racc[b * 64 + j] = 0.0f;
    }
    if (j == 0) { emax[b] = -__builtin_inff(); denom[b] = 0.0f; }
}

__global__ void attn_score(const float* __restrict__ state, const float* __restrict__ hl,
                           const int* __restrict__ batch, float* __restrict__ e,
                           float* __restrict__ emax, int nN) {
    int n = blockIdx.x * blockDim.x + threadIdx.x;
    if (n >= nN) return;
    int b = batch[n];
    const float* x = state + (size_t)n * HID;
    const float* q = hl + b * 64;
    float s = 0.0f;
#pragma unroll 8
    for (int k = 0; k < HID; ++k) s += x[k] * q[k];
    e[n] = s;
    atomicMaxFloat(&emax[b], s);
}

__global__ void attn_soft(const float* __restrict__ e, const float* __restrict__ emax,
                          const int* __restrict__ batch, float* __restrict__ ex,
                          float* __restrict__ denom, int nN) {
    int n = blockIdx.x * blockDim.x + threadIdx.x;
    if (n >= nN) return;
    int b = batch[n];
    float v = __expf(e[n] - emax[b]);
    ex[n] = v;
    atomicAdd(&denom[b], v);
}

__global__ void attn_read(const float* __restrict__ state, const float* __restrict__ ex,
                          const int* __restrict__ batch, float* __restrict__ racc, int nN) {
    int g = blockIdx.x * blockDim.x + threadIdx.x;
    if (g >= nN * HID) return;
    int n = g >> 6, o = g & 63;
    atomicAdd(&racc[batch[n] * HID + o], ex[n] * state[g]);
}

__global__ void qstar_update(float* __restrict__ q_star, const float* __restrict__ hl,
                             const float* __restrict__ racc, const float* __restrict__ denom,
                             int nB) {
    int g = blockIdx.x * blockDim.x + threadIdx.x;
    if (g >= nB * HID) return;
    int b = g >> 6, o = g & 63;
    q_star[b * 128 + o]      = hl[g];
    q_star[b * 128 + 64 + o] = racc[g] / denom[b];
}

__global__ void __launch_bounds__(64)
out_mlp(const float* __restrict__ q_star, const float* __restrict__ w1, const float* __restrict__ b1,
        const float* __restrict__ w2, const float* __restrict__ b2, float* __restrict__ out) {
    __shared__ float red[64];
    const int b = blockIdx.x, o = threadIdx.x;
    const float* q = q_star + b * 128;
    float acc = b1[o];
    for (int k = 0; k < 128; ++k) acc += q[k] * w1[k * 64 + o];
    red[o] = siluf_(acc) * w2[o];
    __syncthreads();
    for (int s = 32; s > 0; s >>= 1) { if (o < s) red[o] += red[o + s]; __syncthreads(); }
    if (o == 0) out[b] = red[0] + b2[0];
}

// ---------------------------------------------------------------- host ----
extern "C" void kernel_launch(void* const* d_in, const int* in_sizes, int n_in,
                              void* d_out, int out_size, void* d_ws, size_t ws_size,
                              hipStream_t stream) {
    const float* x_in   = (const float*)d_in[0];
    const int*   eidx   = (const int*)  d_in[1];
    const float* eattr  = (const float*)d_in[2];
    const int*   batch  = (const int*)  d_in[3];
    const float* fl_w   = (const float*)d_in[4];
    const float* fl_b   = (const float*)d_in[5];
    const float* nn_w1  = (const float*)d_in[6];
    const float* nn_b1  = (const float*)d_in[7];
    const float* nn_w2  = (const float*)d_in[8];
    const float* nn_b2  = (const float*)d_in[9];
    const float* root_w = (const float*)d_in[10];
    const float* conv_b = (const float*)d_in[11];
    const float* gw_ih  = (const float*)d_in[12];
    const float* gw_hh  = (const float*)d_in[13];
    const float* gb_ih  = (const float*)d_in[14];
    const float* gb_hh  = (const float*)d_in[15];
    const float* lw_ih  = (const float*)d_in[16];
    const float* lw_hh  = (const float*)d_in[17];
    const float* lb_ih  = (const float*)d_in[18];
    const float* lb_hh  = (const float*)d_in[19];
    const float* ow1    = (const float*)d_in[20];
    const float* ob1    = (const float*)d_in[21];
    const float* ow2    = (const float*)d_in[22];
    const float* ob2    = (const float*)d_in[23];
    float* out = (float*)d_out;

    const int N = in_sizes[3];          // 10000
    const int E = in_sizes[2] / 4;      // 50000
    const int B = out_size;             // 64
    const int* src = eidx;
    const int* dst = eidx + E;

    // workspace carve-up (256B aligned)
    char* ws = (char*)d_ws;
    size_t off = 0;
    auto alloc = [&](size_t bytes) { size_t o = off; off += (bytes + 255) & ~(size_t)255; return o; };
    float*          state  = (float*)(ws + alloc((size_t)N * HID * 4));
    float*          agg    = (float*)(ws + alloc((size_t)N * HID * 4));
    unsigned short* hidden = (unsigned short*)(ws + alloc((size_t)E * WN * 2));
    unsigned short* w2p    = (unsigned short*)(ws + alloc((size_t)WN * HH * 2));
    float*          q_star = (float*)(ws + alloc((size_t)B * 128 * 4));
    float*          hl     = (float*)(ws + alloc((size_t)B * 64 * 4));
    float*          cl     = (float*)(ws + alloc((size_t)B * 64 * 4));
    float*          racc   = (float*)(ws + alloc((size_t)B * 64 * 4));
    float*          ebuf   = (float*)(ws + alloc((size_t)N * 4));
    float*          exbuf  = (float*)(ws + alloc((size_t)N * 4));
    float*          emax   = (float*)(ws + alloc((size_t)B * 4));
    float*          denom  = (float*)(ws + alloc((size_t)B * 4));
    (void)ws_size; (void)n_in;

    const int T = 256;
    // preprocessing
    first_layer<<<(N * HID + T - 1) / T, T, 0, stream>>>(x_in, fl_w, fl_b, state, N);
    edge_mlp1<<<(E * WN + T - 1) / T, T, 0, stream>>>(eattr, nn_w1, nn_b1, hidden, E);
    pack_w2<<<(WN * HH + T - 1) / T, T, 0, stream>>>(nn_w2, w2p);

    // 4 message-passing layers
    const int nBlk = (E + EPB - 1) / EPB;
    const size_t lds_fused = (size_t)EPB * HID * 4 * 2;               // xs + msgbuf = 16 KB
    const size_t lds_node  = (4096 + 2 * 12288) * sizeof(float);      // ~112 KB
    for (int l = 0; l < 4; ++l) {
        zero_f32<<<(N * HID + T - 1) / T, T, 0, stream>>>(agg, N * HID);
        layer_fused<<<nBlk, T, lds_fused, stream>>>(hidden, w2p, nn_b2, state, src, dst, agg, E);
        node_update<<<(N + 7) / 8, T, lds_node, stream>>>(state, agg, root_w, conv_b,
                                                          gw_ih, gw_hh, gb_ih, gb_hh, N);
    }

    // Set2Set (q_star, hl, cl are contiguous -> one zero pass)
    zero_f32<<<(B * (128 + 64 + 64) + T - 1) / T, T, 0, stream>>>(q_star, B * (128 + 64 + 64));
    for (int s = 0; s < 3; ++s) {
        lstm_step<<<B, 256, 0, stream>>>(q_star, hl, cl, lw_ih, lw_hh, lb_ih, lb_hh,
                                         emax, denom, racc);
        attn_score<<<(N + T - 1) / T, T, 0, stream>>>(state, hl, batch, ebuf, emax, N);
        attn_soft<<<(N + T - 1) / T, T, 0, stream>>>(ebuf, emax, batch, exbuf, denom, N);
        attn_read<<<(N * HID + T - 1) / T, T, 0, stream>>>(state, exbuf, batch, racc, N);
        qstar_update<<<(B * HID + T - 1) / T, T, 0, stream>>>(q_star, hl, racc, denom, B);
    }

    out_mlp<<<B, 64, 0, stream>>>(q_star, ow1, ob1, ow2, ob2, out);
}
